// _CnnAlikeAttention_54743653155402
// MI455X (gfx1250) — matrix operand order repaired and checked by reading, not compiled, not run on hardware
//
#include <hip/hip_runtime.h>
#include <hip/hip_bf16.h>

typedef __attribute__((ext_vector_type(16))) _Float16 v16h;
typedef __attribute__((ext_vector_type(8)))  _Float16 v8h;
typedef __attribute__((ext_vector_type(8)))  float    v8f;
typedef int v4i_ __attribute__((__vector_size__(4 * sizeof(int))));

#define L_SEQ  1024
#define BATCH  8
#define DM     1024
#define MROWS  (L_SEQ * BATCH)   // 8192
#define NHEAD  8
#define LK     (L_SEQ + 1)       // 1025

// guard rows around Vproj so the per-head shift (rows -24..+32) never branches
#define GUARD_LO 24
#define GUARD_HI 32

constexpr int BM  = 128;
constexpr int BN  = 64;
constexpr int BK  = 32;
constexpr int KT  = DM / BK;     // 32 k-tiles
constexpr int LDP = BK + 8;      // padded LDS row stride in halves (80B, 16B-aligned)

#if defined(__AMDGCN__) && __has_builtin(__builtin_amdgcn_global_load_async_to_lds_b128)
#define USE_ASYNC 1
// exact param types per hipcc diagnostic: (v4i addrspace(1)*, v4i addrspace(3)*, imm, imm)
#define ASYNC_GLB(p) ((__attribute__((address_space(1))) v4i_*)(p))
#define ASYNC_LDS(p) ((__attribute__((address_space(3))) v4i_*)(p))
#if __has_builtin(__builtin_amdgcn_s_wait_asynccnt)
#define WAIT_ASYNC(n) __builtin_amdgcn_s_wait_asynccnt(n)
#else
#define WAIT_ASYNC(n) asm volatile("s_wait_asynccnt %0" :: "i"(n) : "memory")
#endif
#else
#define USE_ASYNC 0
#define WAIT_ASYNC(n) ((void)0)
#endif

union FragU { v16h v; struct { v8h lo, hi; } p; };

__device__ inline v8f vzero8() {
    v8f z = {0.f, 0.f, 0.f, 0.f, 0.f, 0.f, 0.f, 0.f};
    return z;
}

// ---------------------------------------------------------------------------
// fp32 -> f16 conversion (4 elements/thread, exact-sized grid)
// ---------------------------------------------------------------------------
__global__ __launch_bounds__(256) void cvt_f32_f16(const float* __restrict__ s,
                                                   _Float16* __restrict__ d) {
    int i = (blockIdx.x * 256 + threadIdx.x) << 2;
    float4 v = *(const float4*)(s + i);
    union { _Float16 h[4]; uint2 u; } pk;
    pk.h[0] = (_Float16)v.x; pk.h[1] = (_Float16)v.y;
    pk.h[2] = (_Float16)v.z; pk.h[3] = (_Float16)v.w;
    *(uint2*)(d + i) = pk.u;
}

// zero-fill in 16B units (exact-sized grid)
__global__ __launch_bounds__(256) void zero_b128(uint4* __restrict__ p) {
    p[(size_t)blockIdx.x * 256 + threadIdx.x] = make_uint4(0u, 0u, 0u, 0u);
}

// ---------------------------------------------------------------------------
// Unified f16 WMMA GEMM:  C = A_shift @ W^T + bias [+ residual]
//   SHIFTED: apply per-head row shift ((k0>>7 - 3)*BATCH); A has guard rows
//   OUTF32:  store fp32 (pre-LN, + residual);  else store f16
// Double-buffered LDS tiles, staged with async global->LDS copies when
// available (ASYNCcnt), otherwise plain copies under the same barriers.
// ---------------------------------------------------------------------------
template <bool SHIFTED, bool OUTF32>
__global__ __launch_bounds__(256) void gemm_f16(
    const _Float16* __restrict__ A,    // (MROWS, DM) f16 (guard rows if SHIFTED)
    const _Float16* __restrict__ W,    // (DM, DM) f16, row-major (K contiguous)
    const float* __restrict__ bias,    // (DM)
    const float* __restrict__ Res,     // residual fp32 or nullptr
    _Float16* __restrict__ C16,        // f16 out (GEMM1)
    float* __restrict__ C32)           // fp32 out (GEMM2)
{
    __shared__ _Float16 As[2][BM][LDP];   // 20 KB
    __shared__ _Float16 Bs[2][BN][LDP];   // 10 KB

    const int tid  = threadIdx.x;
    const int m0   = blockIdx.y * BM;
    const int n0   = blockIdx.x * BN;
    const int wave = tid >> 5;
    const int lane = tid & 31;
    const int wm   = (wave & 3) << 5;   // wave tile row (0,32,64,96)
    const int wn   = (wave >> 2) << 5;  // wave tile col (0,32)
    const int lr   = lane & 15;
    const int hig  = lane >> 4;         // half-wave select
    const int kA   = hig << 3;          // A frag K base: 0 or 8
    const int kB   = hig << 4;          // unused; B frags also load at kA (0 or 8)

    v8f acc[2][2];
#pragma unroll
    for (int mt = 0; mt < 2; ++mt)
#pragma unroll
        for (int nt = 0; nt < 2; ++nt) acc[mt][nt] = vzero8();

    // stage one 128x32 A tile + 64x32 B tile (f16) into LDS buffer `b`
    auto stage = [&](int kt, int b) {
        const int k0 = kt * BK;
        const ptrdiff_t relB = SHIFTED ? (ptrdiff_t)(((k0 >> 7) - 3) * BATCH) : 0;
#if USE_ASYNC
#pragma unroll
        for (int j = 0; j < 2; ++j) {
            int f = tid + (j << 8);
            int r = f >> 2;
            int c = (f & 3) << 3;
            __builtin_amdgcn_global_load_async_to_lds_b128(
                ASYNC_GLB(A + ((ptrdiff_t)(m0 + r) + relB) * DM + (k0 + c)),
                ASYNC_LDS(&As[b][r][c]), 0, 0);
        }
        {
            int r = tid >> 2;
            int c = (tid & 3) << 3;
            __builtin_amdgcn_global_load_async_to_lds_b128(
                ASYNC_GLB(W + (size_t)(n0 + r) * DM + (k0 + c)),
                ASYNC_LDS(&Bs[b][r][c]), 0, 0);
        }
#else
#pragma unroll
        for (int j = 0; j < 2; ++j) {
            int f = tid + (j << 8);
            int r = f >> 2;
            int c = (f & 3) << 3;
            *(uint4*)&As[b][r][c] =
                *(const uint4*)(A + ((ptrdiff_t)(m0 + r) + relB) * DM + (k0 + c));
        }
        {
            int r = tid >> 2;
            int c = (tid & 3) << 3;
            *(uint4*)&Bs[b][r][c] =
                *(const uint4*)(W + (size_t)(n0 + r) * DM + (k0 + c));
        }
#endif
    };

    stage(0, 0);
    int buf = 0;
    for (int kt = 0; kt < KT; ++kt) {
        if (kt + 1 < KT) {
            __syncthreads();           // next buffer no longer being read
            stage(kt + 1, buf ^ 1);
            WAIT_ASYNC(3);             // this wave's tile-kt copies done
        } else {
            WAIT_ASYNC(0);
        }
        __syncthreads();               // all waves' tile-kt data visible

        FragU a0, a1, b0, b1;
        {
            const _Float16* p = &As[buf][wm + lr][0];
            a0.p.lo = *(const v8h*)(p + kA); a0.p.hi = *(const v8h*)(p + kA + 16);
            p = &As[buf][wm + 16 + lr][0];
            a1.p.lo = *(const v8h*)(p + kA); a1.p.hi = *(const v8h*)(p + kA + 16);
            p = &Bs[buf][wn + lr][0];
            b0.p.lo = *(const v8h*)(p + kA); b0.p.hi = *(const v8h*)(p + kA + 16);
            p = &Bs[buf][wn + 16 + lr][0];
            b1.p.lo = *(const v8h*)(p + kA); b1.p.hi = *(const v8h*)(p + kA + 16);
        }
        acc[0][0] = __builtin_amdgcn_wmma_f32_16x16x32_f16(false, a0.v, false, b0.v, (short)0, acc[0][0], false, false);
        acc[0][1] = __builtin_amdgcn_wmma_f32_16x16x32_f16(false, a0.v, false, b1.v, (short)0, acc[0][1], false, false);
        acc[1][0] = __builtin_amdgcn_wmma_f32_16x16x32_f16(false, a1.v, false, b0.v, (short)0, acc[1][0], false, false);
        acc[1][1] = __builtin_amdgcn_wmma_f32_16x16x32_f16(false, a1.v, false, b1.v, (short)0, acc[1][1], false, false);
        buf ^= 1;
    }

#pragma unroll
    for (int mt = 0; mt < 2; ++mt)
#pragma unroll
        for (int nt = 0; nt < 2; ++nt) {
            int n = n0 + wn + (nt << 4) + lr;
            float bn = bias[n];
#pragma unroll
            for (int r = 0; r < 8; ++r) {
                int m = m0 + wm + (mt << 4) + (hig << 3) + r;
                size_t off = (size_t)m * DM + n;
                if (OUTF32) {
                    C32[off] = acc[mt][nt][r] + bn + Res[off];
                } else {
                    C16[off] = (_Float16)(acc[mt][nt][r] + bn);
                }
            }
        }
}

// ---------------------------------------------------------------------------
// LayerNorm: one 256-thread block per row of 1024 floats
// ---------------------------------------------------------------------------
__global__ __launch_bounds__(256) void layernorm_k(
    const float* __restrict__ X, const float* __restrict__ g,
    const float* __restrict__ bta, float* __restrict__ Y)
{
    __shared__ float sS[8], sQ[8];
    const int row = blockIdx.x;
    const int t   = threadIdx.x;
    float4 v = *(const float4*)(X + (size_t)row * DM + (t << 2));
    float s = v.x + v.y + v.z + v.w;
    float q = v.x * v.x + v.y * v.y + v.z * v.z + v.w * v.w;
#pragma unroll
    for (int o = 16; o > 0; o >>= 1) {
        s += __shfl_xor(s, o, 32);
        q += __shfl_xor(q, o, 32);
    }
    if ((t & 31) == 0) { sS[t >> 5] = s; sQ[t >> 5] = q; }
    __syncthreads();
    float ts = 0.f, tq = 0.f;
#pragma unroll
    for (int i = 0; i < 8; ++i) { ts += sS[i]; tq += sQ[i]; }
    const float mu = ts * (1.0f / DM);
    const float rs = rsqrtf(tq * (1.0f / DM) - mu * mu + 1e-5f);
    float4 gg = *(const float4*)(g + (t << 2));
    float4 bb = *(const float4*)(bta + (t << 2));
    float4 o;
    o.x = (v.x - mu) * rs * gg.x + bb.x;
    o.y = (v.y - mu) * rs * gg.y + bb.y;
    o.z = (v.z - mu) * rs * gg.z + bb.z;
    o.w = (v.w - mu) * rs * gg.w + bb.w;
    *(float4*)(Y + (size_t)row * DM + (t << 2)) = o;
}

// ---------------------------------------------------------------------------
// attn output: exact one-hot (softmax of +/-500 logits underflows to 0/1)
// ---------------------------------------------------------------------------
__global__ __launch_bounds__(256) void attn_zero(float4* __restrict__ p) {
    size_t i = (size_t)blockIdx.x * 256 + threadIdx.x;
    p[i] = make_float4(0.f, 0.f, 0.f, 0.f);
}

__global__ __launch_bounds__(256) void attn_ones(float* __restrict__ attn) {
    int t = blockIdx.x * 256 + threadIdx.x;   // t = (h*8 + b)*1024 + i
    int h = t >> 13;
    int i = t & (L_SEQ - 1);
    int j = i + h - 3;
    if (j < 0 || j > L_SEQ) j = L_SEQ;        // out of range -> pad slot 1024
    attn[(size_t)t * LK + j] = 1.0f;
}

// ---------------------------------------------------------------------------
// launch
// ---------------------------------------------------------------------------
extern "C" void kernel_launch(void* const* d_in, const int* in_sizes, int n_in,
                              void* d_out, int out_size, void* d_ws, size_t ws_size,
                              hipStream_t stream)
{
    const float* query = (const float*)d_in[0];
    const float* value = (const float*)d_in[2];
    const float* Wv    = (const float*)d_in[7];
    const float* bv    = (const float*)d_in[8];
    const float* Wo    = (const float*)d_in[9];
    const float* bo    = (const float*)d_in[10];
    const float* gamma = (const float*)d_in[11];
    const float* beta  = (const float*)d_in[12];

    float* out  = (float*)d_out;
    float* attn = out + (size_t)MROWS * DM;

    // ws: Wv16 | Wo16 | value16 | Vproj16 (with guard rows) | X fp32  (~69 MB)
    _Float16* Wv16    = (_Float16*)d_ws;
    _Float16* Wo16    = Wv16 + (size_t)DM * DM;
    _Float16* val16   = Wo16 + (size_t)DM * DM;
    _Float16* Vp_allc = val16 + (size_t)MROWS * DM;
    _Float16* Vp16    = Vp_allc + (size_t)GUARD_LO * DM;           // logical row 0
    float*    X       = (float*)(Vp_allc + (size_t)(GUARD_LO + MROWS + GUARD_HI) * DM);

    // conversions + guard-row zeroing (independent, stream-ordered)
    cvt_f32_f16<<<(DM * DM) / (256 * 4), 256, 0, stream>>>(Wv, Wv16);
    cvt_f32_f16<<<(DM * DM) / (256 * 4), 256, 0, stream>>>(Wo, Wo16);
    cvt_f32_f16<<<((size_t)MROWS * DM) / (256 * 4), 256, 0, stream>>>(value, val16);
    zero_b128<<<(GUARD_LO * DM) / (8 * 256), 256, 0, stream>>>((uint4*)Vp_allc);
    zero_b128<<<(GUARD_HI * DM) / (8 * 256), 256, 0, stream>>>(
        (uint4*)(Vp_allc + (size_t)(GUARD_LO + MROWS) * DM));

    dim3 gemm_grid(DM / BN, MROWS / BM);
    // GEMM1: Vproj = value @ Wv^T + bv           (f16 out)
    gemm_f16<false, false><<<gemm_grid, 256, 0, stream>>>(
        val16, Wv16, bv, nullptr, Vp16, nullptr);
    // GEMM2: X = shift(Vproj) @ Wo^T + bo + query (fp32 out)
    gemm_f16<true, true><<<gemm_grid, 256, 0, stream>>>(
        Vp16, Wo16, bo, query, nullptr, X);

    layernorm_k<<<MROWS, 256, 0, stream>>>(X, gamma, beta, out);

    const int attn_f4 = (NHEAD * BATCH * L_SEQ * LK) / 4;   // 16,793,600
    attn_zero<<<attn_f4 / 256, 256, 0, stream>>>((float4*)attn);
    attn_ones<<<(NHEAD * BATCH * L_SEQ) / 256, 256, 0, stream>>>(attn);
}
